// CentroidSeparationLoss_32289564131920
// MI455X (gfx1250) — matrix operand order repaired and checked by reading, not compiled, not run on hardware
//
#include <hip/hip_runtime.h>
#include <hip/hip_bf16.h>

typedef __attribute__((ext_vector_type(16))) __bf16 v16bf;
typedef __attribute__((ext_vector_type(8)))  float  v8f;
typedef __attribute__((ext_vector_type(4)))  int    v4i;

#define NUM_CLASSES 64
#define FEAT_DIM    128
#define CHUNK_ROWS  32          // rows per pipelined chunk (one K=32 WMMA step)
#define MAIN_GRID   640

// d_ws layout (dwords): [0..8191] sums f32[64][128], [8192..8255] counts u32[64], [8256] sumsq f32
#define WS_WORDS (NUM_CLASSES * FEAT_DIM + NUM_CLASSES + 1)

#if defined(__gfx1250__) && __has_builtin(__builtin_amdgcn_global_load_async_to_lds_b128)
#define ASYNC_OK 1
typedef __attribute__((address_space(1))) v4i g_v4i;   // prints as "__device__" in diags
typedef __attribute__((address_space(3))) v4i l_v4i;   // LDS
#define AS_G(p) ((g_v4i*)(uintptr_t)(p))
#define AS_L(p) ((l_v4i*)(uintptr_t)(p))
#else
#define ASYNC_OK 0
#endif

__device__ __forceinline__ void wait_async_le4() {
#if ASYNC_OK
#if __has_builtin(__builtin_amdgcn_s_wait_asynccnt)
    __builtin_amdgcn_s_wait_asynccnt(4);
#else
    asm volatile("s_wait_asynccnt 4" ::: "memory");
#endif
#endif
}
__device__ __forceinline__ void wait_async_le0() {
#if ASYNC_OK
#if __has_builtin(__builtin_amdgcn_s_wait_asynccnt)
    __builtin_amdgcn_s_wait_asynccnt(0);
#else
    asm volatile("s_wait_asynccnt 0" ::: "memory");
#endif
#endif
}

__device__ __forceinline__ unsigned f32_to_bf16_bits(float f) {
    unsigned u = __float_as_uint(f);
    return (u + 0x7FFFu + ((u >> 16) & 1u)) >> 16;   // round-to-nearest-even
}

__global__ __launch_bounds__(256) void init_ws(unsigned* ws, int nwords) {
    int i = blockIdx.x * 256 + threadIdx.x;
    if (i < nwords) ws[i] = 0u;
}

// ---------------------------------------------------------------------------
// Stage 1: one streaming pass. sums[C][D] via one-hot WMMA GEMM (bf16 in /
// f32 accum). Feature tiles arrive via async global->LDS DMA, double
// buffered; one-hot A words built once per workgroup in LDS and shared by
// all 8 waves. counts via LDS histogram, sumsq in full f32.
// ---------------------------------------------------------------------------
__global__ __launch_bounds__(256) void segment_sum_wmma(
        const float* __restrict__ feats, const int* __restrict__ tgts,
        int nchunks, float* __restrict__ sums, unsigned* __restrict__ counts,
        float* __restrict__ sumsq_g) {
    __shared__ __align__(16) float sfeat[2][CHUNK_ROWS * FEAT_DIM];  // 2 x 16 KB
    __shared__ int      stgt[CHUNK_ROWS];
    __shared__ unsigned sA[4 * 8 * 32];          // packed one-hot A words, 4 KB
    __shared__ unsigned shist[NUM_CLASSES];
    __shared__ float    sred[256];

    const int tid  = threadIdx.x;
    const int lane = tid & 31;
    const int wave = tid >> 5;
    const int n    = lane & 15;        // N (feature col in tile) / M (class row in tile)
    const int h    = lane >> 4;        // lane half
    const int col  = wave * 16 + n;    // this wave's feature column

    if (tid < NUM_CLASSES) shist[tid] = 0u;

    v8f   acc[4] = {};                 // 4 class tiles x (16x16 f32)
    float sumsq  = 0.0f;

#if !ASYNC_OK
    float4 freg[2][4];
#endif
    int treg[2] = {0, 0};
    const int g = gridDim.x;

    auto issue = [&](int ck, int buf) {
#if ASYNC_OK
        const float* gsrc = feats + (size_t)ck * CHUNK_ROWS * FEAT_DIM;
        float*       ldst = &sfeat[buf][0];
#pragma unroll
        for (int k = 0; k < 4; ++k)
            __builtin_amdgcn_global_load_async_to_lds_b128(
                AS_G(gsrc + 4 * (tid + 256 * k)),
                AS_L(ldst + 4 * (tid + 256 * k)), 0, 0);
#else
        const float4* src = (const float4*)(feats + (size_t)ck * CHUNK_ROWS * FEAT_DIM);
#pragma unroll
        for (int k = 0; k < 4; ++k) freg[buf][k] = src[tid + 256 * k];
#endif
    };

    int chunk = blockIdx.x;
    if (chunk < nchunks) {
        issue(chunk, 0);
        if (tid < CHUNK_ROWS) treg[0] = tgts[chunk * CHUNK_ROWS + tid];
    }
    if (chunk + g < nchunks) {
        issue(chunk + g, 1);
        if (tid < CHUNK_ROWS) treg[1] = tgts[(chunk + g) * CHUNK_ROWS + tid];
    }

    int parity = 0;
    while (chunk < nchunks) {
        // retire DMA for this chunk (in-order: trailing 4 belong to chunk+g if issued)
        if (chunk + g < nchunks) wait_async_le4(); else wait_async_le0();
#if !ASYNC_OK
        {
            float4* dst = (float4*)&sfeat[parity][0];
#pragma unroll
            for (int k = 0; k < 4; ++k) dst[tid + 256 * k] = freg[parity][k];
        }
#endif
        if (tid < CHUNK_ROWS) {
            stgt[tid] = treg[parity];
            atomicAdd(&shist[treg[parity] & (NUM_CLASSES - 1)], 1u);
        }
        __syncthreads();                       // features + targets visible

        // build the 1024 packed one-hot A words once per workgroup (4/thread)
#pragma unroll
        for (int w = 0; w < 4; ++w) {
            int widx = w * 256 + tid;          // = ct*256 + j*32 + l
            int l  = widx & 31;
            int j  = (widx >> 5) & 7;
            int ct = widx >> 8;
            int m  = l & 15, hh = l >> 4;
            int k0 = 16 * hh + 2 * j;
            int t0 = stgt[k0], t1 = stgt[k0 + 1];
            int cls = ct * 16 + m;
            sA[widx] = ((t0 == cls) ? 0x3F80u : 0u) |          // bf16 1.0 lo
                       ((t1 == cls) ? 0x3F800000u : 0u);       // bf16 1.0 hi
        }
        __syncthreads();                       // sA ready

        // B tile: 32x16 bf16, B[k][n] = feat[k][col]; exact f32 sumsq on the way
        union { unsigned u[8]; v16bf v; } Bm;
#pragma unroll
        for (int j = 0; j < 8; ++j) {
            float f0 = sfeat[parity][(16 * h + 2 * j) * FEAT_DIM + col];
            float f1 = sfeat[parity][(16 * h + 2 * j + 1) * FEAT_DIM + col];
            sumsq += f0 * f0 + f1 * f1;
            Bm.u[j] = f32_to_bf16_bits(f0) | (f32_to_bf16_bits(f1) << 16);
        }
#pragma unroll
        for (int ct = 0; ct < 4; ++ct) {
            union { unsigned u[8]; v16bf v; } Am;
#pragma unroll
            for (int j = 0; j < 8; ++j) Am.u[j] = sA[(ct * 8 + j) * 32 + lane];
            acc[ct] = __builtin_amdgcn_wmma_f32_16x16x32_bf16(
                false, Am.v, false, Bm.v, (short)0, acc[ct], false, false);
        }
        __syncthreads();                       // all waves done with this buffer

        int nc = chunk + 2 * g;                // refill the buffer two chunks ahead
        if (nc < nchunks) {
            issue(nc, parity);
            if (tid < CHUNK_ROWS) treg[parity] = tgts[nc * CHUNK_ROWS + tid];
        }
        chunk += g;
        parity ^= 1;
    }

    // flush accumulators: VGPR r -> class ct*16 + r + 8h, feature col
#pragma unroll
    for (int ct = 0; ct < 4; ++ct)
#pragma unroll
        for (int r = 0; r < 8; ++r)
            atomicAdd(&sums[(ct * 16 + r + 8 * h) * FEAT_DIM + col], acc[ct][r]);

    __syncthreads();
    sred[tid] = sumsq;
    __syncthreads();
    for (int s = 128; s > 0; s >>= 1) {
        if (tid < s) sred[tid] += sred[tid + s];
        __syncthreads();
    }
    if (tid == 0) atomicAdd(sumsq_g, sred[0]);
    if (tid < NUM_CLASSES) atomicAdd(&counts[tid], shist[tid]);
}

// Scalar fallback for a non-multiple-of-32 batch tail (not used at B=1e6).
__global__ void tail_rows(const float* __restrict__ feats, const int* __restrict__ tgts,
                          int start, int btot, float* sums, unsigned* counts, float* sumsq_g) {
    int r = start + blockIdx.x * 64 + threadIdx.x;
    if (r >= btot) return;
    int t = tgts[r] & (NUM_CLASSES - 1);
    atomicAdd(&counts[t], 1u);
    float sq = 0.0f;
    for (int d = 0; d < FEAT_DIM; ++d) {
        float f = feats[(size_t)r * FEAT_DIM + d];
        sq += f * f;
        atomicAdd(&sums[t * FEAT_DIM + d], f);
    }
    atomicAdd(sumsq_g, sq);
}

// ---------------------------------------------------------------------------
// Finalize: centers, intra (closed form), Gram = centers x centers^T via
// WMMA bf16, hinge-weighted inter, scalar out.
// ---------------------------------------------------------------------------
__global__ __launch_bounds__(256) void finalize_loss(
        const float* __restrict__ sums, const unsigned* __restrict__ counts,
        const float* __restrict__ sumsq_g, float* __restrict__ out, int btot) {
    __shared__ __align__(16) float scent[NUM_CLASSES * FEAT_DIM];  // 32 KB
    __shared__ float sG[NUM_CLASSES * NUM_CLASSES];                // 16 KB
    __shared__ float sred[256];
    __shared__ float scnt[NUM_CLASSES];
    __shared__ float sintra;

    const int tid  = threadIdx.x;
    const int lane = tid & 31;
    const int wave = tid >> 5;
    const int n    = lane & 15;
    const int h    = lane >> 4;

    if (tid < NUM_CLASSES) {
        unsigned c = counts[tid];
        scnt[tid] = (c > 0u) ? (float)c : 1.0f;
    }
    __syncthreads();

    float part = 0.0f;
    for (int i = tid; i < NUM_CLASSES * FEAT_DIM; i += 256) {
        float s  = sums[i];
        float nc = scnt[i >> 7];
        scent[i] = s / nc;
        part += s * s / nc;
    }
    __syncthreads();
    sred[tid] = part;
    __syncthreads();
    for (int s = 128; s > 0; s >>= 1) {
        if (tid < s) sred[tid] += sred[tid + s];
        __syncthreads();
    }
    if (tid == 0) sintra = (sumsq_g[0] - sred[0]) / (float)btot;
    __syncthreads();

    // G = centers x centers^T: 4x4 tiles of 16x16, 2 tiles/wave, K=128 in 4 steps
    for (int tt = wave * 2; tt < wave * 2 + 2; ++tt) {
        const int ti = tt >> 2, tj = tt & 3;
        v8f acc = {};
        for (int ks = 0; ks < 4; ++ks) {
            union { unsigned u[8]; v16bf v; } Am, Bm;
#pragma unroll
            for (int j = 0; j < 8; ++j) {
                int ka = ks * 32 + 16 * h + 2 * j;
                float a0 = scent[(ti * 16 + n) * FEAT_DIM + ka];
                float a1 = scent[(ti * 16 + n) * FEAT_DIM + ka + 1];
                Am.u[j] = f32_to_bf16_bits(a0) | (f32_to_bf16_bits(a1) << 16);
                int kb = ks * 32 + 16 * h + 2 * j;
                float b0 = scent[(tj * 16 + n) * FEAT_DIM + kb];
                float b1 = scent[(tj * 16 + n) * FEAT_DIM + kb + 1];
                Bm.u[j] = f32_to_bf16_bits(b0) | (f32_to_bf16_bits(b1) << 16);
            }
            acc = __builtin_amdgcn_wmma_f32_16x16x32_bf16(
                false, Am.v, false, Bm.v, (short)0, acc, false, false);
        }
#pragma unroll
        for (int r = 0; r < 8; ++r)
            sG[(ti * 16 + r + 8 * h) * NUM_CLASSES + tj * 16 + n] = acc[r];
    }
    __syncthreads();

    float hs = 0.0f;
    for (int e = tid; e < NUM_CLASSES * NUM_CLASSES; e += 256) {
        int i = e >> 6, j = e & 63;
        if (i < j) {
            float d2 = sG[i * NUM_CLASSES + i] + sG[j * NUM_CLASSES + j]
                     - 2.0f * sG[i * NUM_CLASSES + j];
            float hg = fmaxf(2.0f - d2, 0.0f);            // MARGIN = 2.0
            float w  = (i == 1 && j == 2) ? 2.0f : 1.0f;  // weights[1,2] = 2
            hs += w * hg;
        }
    }
    sred[tid] = hs;
    __syncthreads();
    for (int s = 128; s > 0; s >>= 1) {
        if (tid < s) sred[tid] += sred[tid + s];
        __syncthreads();
    }
    if (tid == 0)
        out[0] = sintra + sred[0] * (1.0f / 2016.0f);     // n_pairs = 64*63/2
}

extern "C" void kernel_launch(void* const* d_in, const int* in_sizes, int n_in,
                              void* d_out, int out_size, void* d_ws, size_t ws_size,
                              hipStream_t stream) {
    (void)n_in; (void)out_size; (void)ws_size;
    const float* feats = (const float*)d_in[0];
    const int*   tgts  = (const int*)d_in[1];
    const int    btot  = in_sizes[0] / FEAT_DIM;

    float*    sums   = (float*)d_ws;
    unsigned* counts = (unsigned*)((char*)d_ws + NUM_CLASSES * FEAT_DIM * sizeof(float));
    float*    sumsq  = (float*)((char*)d_ws + NUM_CLASSES * FEAT_DIM * sizeof(float)
                                            + NUM_CLASSES * sizeof(unsigned));

    init_ws<<<(WS_WORDS + 255) / 256, 256, 0, stream>>>((unsigned*)d_ws, WS_WORDS);

    const int nchunks = btot / CHUNK_ROWS;
    if (nchunks > 0) {
        int grid = MAIN_GRID < nchunks ? MAIN_GRID : nchunks;
        segment_sum_wmma<<<grid, 256, 0, stream>>>(feats, tgts, nchunks, sums, counts, sumsq);
    }
    const int rem_start = nchunks * CHUNK_ROWS;
    const int rem = btot - rem_start;
    if (rem > 0)
        tail_rows<<<(rem + 63) / 64, 64, 0, stream>>>(feats, tgts, rem_start, btot,
                                                      sums, counts, sumsq);

    finalize_loss<<<1, 256, 0, stream>>>(sums, counts, sumsq, (float*)d_out, btot);
}
